// GCN_mgaev3_35141422416147
// MI455X (gfx1250) — matrix operand order repaired and checked by reading, not compiled, not run on hardware
//
#include <hip/hip_runtime.h>
#include <hip/hip_bf16.h>

// ---------------------------------------------------------------------------
// Co-attention GCN (2 layers) on B=512 graphs of 90 nodes each, for gfx1250.
// All matmuls routed through v_wmma_f32_16x16x32_f16 (wave32 WMMA).
// Sparse message passing converted to dense per-subject 96x96 \hat{A} GEMMs.
// Weights stored transposed so every B-fragment is two contiguous 16B runs/lane;
// batched GEMM stages A + transposed B slab in LDS (66KB/WG of the 320KB WGP).
// ---------------------------------------------------------------------------

typedef _Float16 half_t;
typedef __attribute__((ext_vector_type(16))) _Float16 v16h;
typedef __attribute__((ext_vector_type(8)))  _Float16 v8h;
typedef __attribute__((ext_vector_type(8)))  float    v8f;

#define NODES   90
#define NPAD    96
#define NSUBJ   512
#define NTOT    (NSUBJ * NODES)      // 46080 (multiple of 64)
#define NSLACK  (NTOT + 16)          // slack rows for cross-subject tile overreads
#define EPER    2700
#define DH      256                  // hidden dim

// ---------------- WMMA fragment loaders (wave32, 16x16x32 f16) -------------
// A-matrix 16x32 f16: lanes 0-15 row M=lane, K in {k0..k0+7, k0+16..k0+23};
// lanes 16-31 row M=lane-16, K base +8. v16h elem j <-> VGPR j/2 half j%2.
// Loads are two contiguous 16B chunks (b128 each); all call sites 16B aligned.
__device__ __forceinline__ v16h load_a_frag(const half_t* A, int lda, int row0,
                                            int k0, int lane) {
  const int r  = row0 + (lane & 15);
  const int kb = k0 + ((lane >> 4) << 3);          // 0 or 8
  const half_t* p = A + (size_t)r * lda + kb;
  v8h lo = *(const v8h*)p;
  v8h hi = *(const v8h*)(p + 16);
  v16h a;
#pragma unroll
  for (int i = 0; i < 8; ++i) { a[i] = lo[i]; a[8 + i] = hi[i]; }
  return a;
}

// B-fragment from a TRANSPOSED operand BT[n][k] (row-major, leading dim ldt):
// logical B[k,n] = BT[n,k]. Per lane: two 8-element (16B) runs, K layout as A.
// v16h elem j = BT[n, kb+j] (j<8), BT[n, kb+16+(j-8)] (j>=8), kb = k0 + 8*(lane/16).
__device__ __forceinline__ v16h load_bt_frag(const half_t* BT, int ldt, int k0,
                                             int col0, int lane) {
  const int n  = col0 + (lane & 15);
  const int kb = k0 + ((lane >> 4) << 3);          // 0 or 8
  const half_t* p = BT + (size_t)n * ldt + kb;
  v8h lo = *(const v8h*)p;
  v8h hi = *(const v8h*)(p + 16);
  v16h b;
#pragma unroll
  for (int i = 0; i < 8; ++i) { b[i] = lo[i]; b[8 + i] = hi[i]; }
  return b;
}

#define WMMA_F16(acc, a, b) \
  __builtin_amdgcn_wmma_f32_16x16x32_f16(false, (a), false, (b), (short)0, (acc), false, false)

// C/D 16x16 f32 layout: lanes 0-15 col N=lane, VGPR v -> M=v; lanes 16-31 -> M=v+8.

// ---------------- f32 -> f16 conversion with zero padding ------------------
__global__ __launch_bounds__(256) void k_cvt(const float* __restrict__ src,
                                             half_t* __restrict__ dst,
                                             int rs, int cs, int rd, int cd) {
  size_t i = (size_t)blockIdx.x * blockDim.x + threadIdx.x;
  size_t tot = (size_t)rd * cd;
  if (i >= tot) return;
  int r = (int)(i / cd), c = (int)(i % cd);
  float v = (r < rs && c < cs) ? src[(size_t)r * cs + c] : 0.0f;
  dst[i] = (half_t)v;
}

// transposed+padded: dst[c*rk + r] = src[r*cs + c] (r<rs), zero pad to rk rows
__global__ __launch_bounds__(256) void k_cvtT(const float* __restrict__ src,
                                              half_t* __restrict__ dst,
                                              int rs, int cs, int rk) {
  size_t i = (size_t)blockIdx.x * blockDim.x + threadIdx.x;
  size_t tot = (size_t)cs * rk;
  if (i >= tot) return;
  int c = (int)(i / rk), r = (int)(i % rk);
  float v = (r < rs) ? src[(size_t)r * cs + c] : 0.0f;
  dst[i] = (half_t)v;
}

// ---------------- dense normalized adjacency per subject -------------------
// Ahat[d][s] = dinv[d]*dinv[s]*count(s->d), padded to 96x96 f16, zeros outside.
__global__ __launch_bounds__(256) void k_build_adj(const long long* __restrict__ src,
                                                   const long long* __restrict__ dst,
                                                   half_t* __restrict__ Ahat) {
  __shared__ float cnt[NODES * NODES];
  __shared__ float dinv[NODES];
  const int b = blockIdx.x;
  for (int i = threadIdx.x; i < NODES * NODES; i += blockDim.x) cnt[i] = 0.0f;
  __syncthreads();
  const long long off = (long long)b * NODES;
  const int e0 = b * EPER;
  for (int e = threadIdx.x; e < EPER; e += blockDim.x) {
    int s = (int)(src[e0 + e] - off);
    int d = (int)(dst[e0 + e] - off);
    atomicAdd(&cnt[d * NODES + s], 1.0f);
  }
  __syncthreads();
  if (threadIdx.x < NODES) {
    int d = threadIdx.x; float deg = 0.0f;
    for (int s = 0; s < NODES; ++s) deg += cnt[d * NODES + s];
    dinv[d] = (deg > 0.0f) ? rsqrtf(deg) : 0.0f;
  }
  __syncthreads();
  half_t* out = Ahat + (size_t)b * NPAD * NPAD;
  for (int i = threadIdx.x; i < NPAD * NPAD; i += blockDim.x) {
    int d = i / NPAD, s = i % NPAD;
    float v = (d < NODES && s < NODES) ? dinv[d] * dinv[s] * cnt[d * NODES + s] : 0.0f;
    out[i] = (half_t)v;
  }
}

// ---------------- tiled WMMA GEMM: Ch = A(MxK) @ BT^T(KxNcols) -------------
// BT is the transposed weight [Ncols x K]. Block = 256 threads = 8 waves in a
// 4x2 layout; each wave computes a 16x64 strip (4 accumulators) so one A
// fragment feeds 4 WMMAs. grid = (M/64, Ncols/128).
__global__ __launch_bounds__(256) void k_gemm(const half_t* __restrict__ A, int lda,
                                              const half_t* __restrict__ BT,
                                              half_t* __restrict__ Ch, int ldch,
                                              int K) {
  const int lane = threadIdx.x & 31;
  const int w    = threadIdx.x >> 5;
  const int row0 = blockIdx.x * 64 + (w & 3) * 16;
  const int colg = blockIdx.y * 128 + (w >> 2) * 64;
  v8f acc0 = {}, acc1 = {}, acc2 = {}, acc3 = {};
  for (int k = 0; k < K; k += 32) {
    if (k + 32 < K)
      __builtin_prefetch(A + (size_t)(row0 + (lane & 15)) * lda + k + 32, 0, 1);
    v16h af = load_a_frag(A, lda, row0, k, lane);
    v16h b0 = load_bt_frag(BT, K, k, colg,      lane);
    v16h b1 = load_bt_frag(BT, K, k, colg + 16, lane);
    v16h b2 = load_bt_frag(BT, K, k, colg + 32, lane);
    v16h b3 = load_bt_frag(BT, K, k, colg + 48, lane);
    acc0 = WMMA_F16(acc0, af, b0);
    acc1 = WMMA_F16(acc1, af, b1);
    acc2 = WMMA_F16(acc2, af, b2);
    acc3 = WMMA_F16(acc3, af, b3);
  }
  const int m_off = (lane >> 4) << 3;
  const int cc    = lane & 15;
#pragma unroll
  for (int v = 0; v < 8; ++v) {
    half_t* q = Ch + (size_t)(row0 + m_off + v) * ldch + colg + cc;
    q[0]  = (half_t)acc0[v];
    q[16] = (half_t)acc1[v];
    q[32] = (half_t)acc2[v];
    q[48] = (half_t)acc3[v];
  }
}

// ---------------- batched 96x96 @ 96x256 GEMM (agg + co-attn apply) --------
// One block per subject. A (96x96) and the transposed 96x256 feature slab are
// staged in LDS (18KB + 48KB); 8 waves then sweep the 6x16 output tiles with
// pure LDS fragment reads. Store masks rows >= 90.
#define BG_SMEM ((NPAD * NPAD + DH * NPAD) * (int)sizeof(half_t))
__global__ __launch_bounds__(256) void k_bgemm96(const half_t* __restrict__ Abase,
                                                 const half_t* __restrict__ Bbase, int ldb,
                                                 float* __restrict__ Cf, int ldcf,
                                                 half_t* __restrict__ Ch, int ldch,
                                                 const float* __restrict__ bias,
                                                 int relu) {
  extern __shared__ char smem_raw[];
  half_t* As = (half_t*)smem_raw;                          // [96][96]
  half_t* Bs = As + NPAD * NPAD;                           // BT: [256][96]
  const int b    = blockIdx.x;
  const int lane = threadIdx.x & 31;
  const int w    = threadIdx.x >> 5;
  const half_t* Ag = Abase + (size_t)b * NPAD * NPAD;
  const half_t* Bg = Bbase + (size_t)b * NODES * ldb;
  // stage A: contiguous 16B chunks
  for (int i = threadIdx.x; i < (NPAD * NPAD) / 8; i += 256)
    ((v8h*)As)[i] = ((const v8h*)Ag)[i];
  // stage B transposed: coalesced global reads (c fastest), Bs[c][k]
  for (int i = threadIdx.x; i < NPAD * DH; i += 256) {
    int k = i >> 8, c = i & (DH - 1);
    Bs[c * NPAD + k] = (k < NODES) ? Bg[(size_t)k * ldb + c] : (half_t)0.0f;
  }
  __syncthreads();
  for (int t = w; t < 96; t += 8) {
    const int rt = (t >> 4) * 16;            // 6 row tiles
    const int ct = (t & 15) * 16;            // 16 col tiles
    v8f acc = {};
#pragma unroll
    for (int k = 0; k < NPAD; k += 32) {
      v16h af = load_a_frag(As, NPAD, rt, k, lane);
      v16h bf = load_bt_frag(Bs, NPAD, k, ct, lane);
      acc = WMMA_F16(acc, af, bf);
    }
    const int m_off = (lane >> 4) << 3;
    const int col   = ct + (lane & 15);
    const float bv  = bias ? bias[col] : 0.0f;
    float res[8];
#pragma unroll
    for (int v = 0; v < 8; ++v) {
      float x = acc[v] + bv;
      res[v] = relu ? fmaxf(x, 0.0f) : x;
    }
    if (Cf) {
#pragma unroll
      for (int v = 0; v < 8; ++v) {
        int m = rt + m_off + v;
        if (m < NODES) Cf[((size_t)b * NODES + m) * ldcf + col] = res[v];
      }
    }
    if (Ch) {
#pragma unroll
      for (int v = 0; v < 8; ++v) {
        int m = rt + m_off + v;
        if (m < NODES) Ch[((size_t)b * NODES + m) * ldch + col] = (half_t)res[v];
      }
    }
  }
}

// ---------------- co-attention scores: A=tanh(T@F^T), row & col softmax ----
// One block per subject. F^T fragments are contiguous runs of F's rows, so no
// transpose staging is needed. Threads 0-95: row softmax -> P; threads
// 96-191: column softmax -> Pt (f16, zero padded to 96x96).
__global__ __launch_bounds__(256) void k_attn(const half_t* __restrict__ Tbase, int ldt,
                                              const half_t* __restrict__ Fbase, int ldf,
                                              half_t* __restrict__ P,
                                              half_t* __restrict__ Pt) {
  __shared__ float Asm[NPAD][NPAD + 1];
  const int b    = blockIdx.x;
  const int lane = threadIdx.x & 31;
  const int w    = threadIdx.x >> 5;
  const half_t* T = Tbase + (size_t)b * NODES * ldt;
  const half_t* F = Fbase + (size_t)b * NODES * ldf;
  for (int t = w; t < 36; t += 8) {          // 6x6 tiles of the 96x96 scores
    const int rt = (t / 6) * 16;
    const int ct = (t % 6) * 16;
    v8f acc = {};
#pragma unroll
    for (int k = 0; k < DH; k += 32) {
      v16h af = load_a_frag(T, ldt, rt, k, lane);
      v16h bf = load_bt_frag(F, ldf, k, ct, lane);
      acc = WMMA_F16(acc, af, bf);
    }
    const int m_off = (lane >> 4) << 3;
    const int cc    = lane & 15;
#pragma unroll
    for (int v = 0; v < 8; ++v)
      Asm[rt + m_off + v][ct + cc] = tanhf(acc[v]);
  }
  __syncthreads();
  if (threadIdx.x < NPAD) {                  // row softmax -> P[b]
    const int r = threadIdx.x;
    half_t* Prow = P + ((size_t)b * NPAD + r) * NPAD;
    if (r < NODES) {
      float mx = -1e30f;
      for (int c = 0; c < NODES; ++c) mx = fmaxf(mx, Asm[r][c]);
      float s = 0.0f;
      for (int c = 0; c < NODES; ++c) s += __expf(Asm[r][c] - mx);
      float inv = 1.0f / s;
      for (int c = 0; c < NPAD; ++c)
        Prow[c] = (c < NODES) ? (half_t)(__expf(Asm[r][c] - mx) * inv) : (half_t)0.0f;
    } else {
      for (int c = 0; c < NPAD; ++c) Prow[c] = (half_t)0.0f;
    }
  } else if (threadIdx.x < 2 * NPAD) {       // column softmax -> Pt[b]
    const int m = threadIdx.x - NPAD;
    half_t* Prow = Pt + ((size_t)b * NPAD + m) * NPAD;
    if (m < NODES) {
      float mx = -1e30f;
      for (int n = 0; n < NODES; ++n) mx = fmaxf(mx, Asm[n][m]);
      float s = 0.0f;
      for (int n = 0; n < NODES; ++n) s += __expf(Asm[n][m] - mx);
      float inv = 1.0f / s;
      for (int n = 0; n < NPAD; ++n)
        Prow[n] = (n < NODES) ? (half_t)(__expf(Asm[n][m] - mx) * inv) : (half_t)0.0f;
    } else {
      for (int n = 0; n < NPAD; ++n) Prow[n] = (half_t)0.0f;
    }
  }
}

// ---------------------------------------------------------------------------
static inline char* bump(char*& p, size_t bytes) {
  char* r = p;
  p += (bytes + 255) & ~(size_t)255;
  return r;
}

extern "C" void kernel_launch(void* const* d_in, const int* in_sizes, int n_in,
                              void* d_out, int out_size, void* d_ws, size_t ws_size,
                              hipStream_t stream) {
  (void)in_sizes; (void)n_in; (void)out_size; (void)ws_size;
  const float*     x_sc = (const float*)d_in[0];
  const long long* e_sc = (const long long*)d_in[1];
  const float*     x_fc = (const float*)d_in[2];
  const long long* e_fc = (const long long*)d_in[3];
  const float*     W0   = (const float*)d_in[4];
  const float*     b0   = (const float*)d_in[5];
  const float*     W1   = (const float*)d_in[6];
  const float*     b1   = (const float*)d_in[7];
  const float*     Wa   = (const float*)d_in[8];
  float* out = (float*)d_out;

  const size_t EDGES = (size_t)NSUBJ * EPER;
  float* o_x1s = out;
  float* o_x2s = out + (size_t)NTOT * 512;
  float* o_x1f = out + (size_t)2 * NTOT * 512;
  float* o_x2f = out + (size_t)3 * NTOT * 512;

  char* p = (char*)d_ws;
  half_t* XscH  = (half_t*)bump(p, (size_t)NTOT * NPAD * 2);
  half_t* XfcH  = (half_t*)bump(p, (size_t)NTOT * NPAD * 2);
  half_t* W0T   = (half_t*)bump(p, (size_t)DH * NPAD * 2);    // [256][96]
  half_t* W1T   = (half_t*)bump(p, (size_t)DH * 512 * 2);     // [256][512]
  half_t* WaT   = (half_t*)bump(p, (size_t)DH * DH * 2);      // [256][256]
  half_t* AhatS = (half_t*)bump(p, (size_t)NSUBJ * NPAD * NPAD * 2);
  half_t* AhatF = (half_t*)bump(p, (size_t)NSUBJ * NPAD * NPAD * 2);
  half_t* HsH   = (half_t*)bump(p, (size_t)NSLACK * DH * 2);  // pre-agg H / T staging
  half_t* HfH   = (half_t*)bump(p, (size_t)NSLACK * DH * 2);
  half_t* PH    = (half_t*)bump(p, (size_t)NSUBJ * NPAD * NPAD * 2);
  half_t* PtH   = (half_t*)bump(p, (size_t)NSUBJ * NPAD * NPAD * 2);
  half_t* X1sH  = (half_t*)bump(p, (size_t)NSLACK * 512 * 2);
  half_t* X1fH  = (half_t*)bump(p, (size_t)NSLACK * 512 * 2);
  half_t* X2sH  = (half_t*)bump(p, (size_t)NSLACK * DH * 2);
  half_t* X2fH  = (half_t*)bump(p, (size_t)NSLACK * DH * 2);

  auto cvt = [&](const float* s, half_t* d, int rs, int cs, int rd, int cd) {
    size_t tot = (size_t)rd * cd;
    k_cvt<<<dim3((unsigned)((tot + 255) / 256)), dim3(256), 0, stream>>>(s, d, rs, cs, rd, cd);
  };
  auto cvtT = [&](const float* s, half_t* d, int rs, int cs, int rk) {
    size_t tot = (size_t)cs * rk;
    k_cvtT<<<dim3((unsigned)((tot + 255) / 256)), dim3(256), 0, stream>>>(s, d, rs, cs, rk);
  };

  // ---- precision staging (weights transposed) ----
  cvt(x_sc, XscH, NTOT, NODES, NTOT, NPAD);
  cvt(x_fc, XfcH, NTOT, NODES, NTOT, NPAD);
  cvtT(W0, W0T, NODES, DH, NPAD);
  cvtT(W1, W1T, 512,   DH, 512);
  cvtT(Wa, WaT, DH,    DH, DH);

  // ---- dense normalized adjacency per subject ----
  k_build_adj<<<NSUBJ, 256, 0, stream>>>(e_sc, e_sc + EDGES, AhatS);
  k_build_adj<<<NSUBJ, 256, 0, stream>>>(e_fc, e_fc + EDGES, AhatF);

  const dim3 gGemm(NTOT / 64, DH / 128);

  // ---- layer 1: H = X @ W0 ----
  k_gemm<<<gGemm, 256, 0, stream>>>(XscH, NPAD, W0T, HsH, DH, NPAD);
  k_gemm<<<gGemm, 256, 0, stream>>>(XfcH, NPAD, W0T, HfH, DH, NPAD);
  // aggregation: relu(Ahat @ H + b0) -> d_out cols [0,256) and f16 X1 buffers
  k_bgemm96<<<NSUBJ, 256, BG_SMEM, stream>>>(AhatS, HsH, DH, o_x1s, 512, X1sH, 512, b0, 1);
  k_bgemm96<<<NSUBJ, 256, BG_SMEM, stream>>>(AhatF, HfH, DH, o_x1f, 512, X1fH, 512, b0, 1);

  // ---- co-attention 1: T = S@Wa; A = tanh(T@F^T); P,Pt; apply ----
  k_gemm<<<gGemm, 256, 0, stream>>>(X1sH, 512, WaT, HsH, DH, DH);
  k_attn<<<NSUBJ, 256, 0, stream>>>(HsH, DH, X1fH, 512, PH, PtH);
  k_bgemm96<<<NSUBJ, 256, BG_SMEM, stream>>>(PH,  X1fH, 512, o_x1s + 256, 512, X1sH + 256, 512, nullptr, 0);
  k_bgemm96<<<NSUBJ, 256, BG_SMEM, stream>>>(PtH, X1sH, 512, o_x1f + 256, 512, X1fH + 256, 512, nullptr, 0);

  // ---- layer 2: H2 = X1 @ W1 (K=512) ----
  k_gemm<<<gGemm, 256, 0, stream>>>(X1sH, 512, W1T, HsH, DH, 512);
  k_gemm<<<gGemm, 256, 0, stream>>>(X1fH, 512, W1T, HfH, DH, 512);
  k_bgemm96<<<NSUBJ, 256, BG_SMEM, stream>>>(AhatS, HsH, DH, o_x2s, 512, X2sH, DH, b1, 1);
  k_bgemm96<<<NSUBJ, 256, BG_SMEM, stream>>>(AhatF, HfH, DH, o_x2f, 512, X2fH, DH, b1, 1);

  // ---- co-attention 2 ----
  k_gemm<<<gGemm, 256, 0, stream>>>(X2sH, DH, WaT, HsH, DH, DH);
  k_attn<<<NSUBJ, 256, 0, stream>>>(HsH, DH, X2fH, DH, PH, PtH);
  k_bgemm96<<<NSUBJ, 256, BG_SMEM, stream>>>(PH,  X2fH, DH, o_x2s + 256, 512, nullptr, 0, nullptr, 0);
  k_bgemm96<<<NSUBJ, 256, BG_SMEM, stream>>>(PtH, X2sH, DH, o_x2f + 256, 512, nullptr, 0, nullptr, 0);
}
